// GVPGraphEncoderHybrid_5592047420123
// MI455X (gfx1250) — compile-verified, matrix operand order checked by reading, not run on hardware
//
#include <hip/hip_runtime.h>
#include <math.h>

typedef __bf16 bf16_t;
typedef __attribute__((ext_vector_type(16))) __bf16 v16bf;
typedef __attribute__((ext_vector_type(8)))  float  v8f;

#define EPSF 1e-8f

// ---------------------------------------------------------------------------
// WMMA helpers (CDNA5 v_wmma_f32_16x16x32_bf16, f32 accumulate)
// ---------------------------------------------------------------------------
__device__ __forceinline__ v8f zero8() {
  v8f z = {0.f, 0.f, 0.f, 0.f, 0.f, 0.f, 0.f, 0.f};
  return z;
}

__device__ __forceinline__ v8f wmma_bf16(v16bf a, v16bf b, v8f c) {
  return __builtin_amdgcn_wmma_f32_16x16x32_bf16(
      /*neg_a=*/false, a, /*neg_b=*/false, b,
      /*c_mod=*/(short)0, c, /*reuse_a=*/false, /*reuse_b=*/false);
}

// Load a 16x32 bf16 fragment from a row-major [16][ld] array (A matrix), or a
// 32x16 B fragment from a pre-transposed weight array Wt[N][ld] (pass row base
// pointer). Per ISA 7.12.2, 16-bit A: lanes 0-15 hold M=0..15; VGPR r holds K
// pairs: r<4 -> K = half*8 + 2r, r>=4 -> K = 16 + half*8 + 2(r-4).
__device__ __forceinline__ v16bf load_frag16(const bf16_t* p, int ld, int kbase,
                                             int lane) {
  const int m = lane & 15, half = lane >> 4;
  v16bf f;
#pragma unroll
  for (int r = 0; r < 8; ++r) {
    const int k = kbase + ((r & 4) << 2) + (half << 3) + ((r & 3) << 1);
    f[2 * r]     = p[m * ld + k];
    f[2 * r + 1] = p[m * ld + k + 1];
  }
  return f;
}

// ---------------------------------------------------------------------------
// Small utility kernels
// ---------------------------------------------------------------------------
__global__ void fill_zero_kernel(float* __restrict__ p, int n) {
  int i = blockIdx.x * blockDim.x + threadIdx.x;
  if (i < n) p[i] = 0.f;
}

__global__ void edge_count_kernel(const int* __restrict__ dst,
                                  float* __restrict__ cnt, int E) {
  int i = blockIdx.x * blockDim.x + threadIdx.x;
  if (i < E) atomicAdd(&cnt[dst[i]], 1.f);
}

__global__ void inv_count_kernel(float* __restrict__ cnt, int n) {
  int i = blockIdx.x * blockDim.x + threadIdx.x;
  if (i < n) cnt[i] = 1.f / fmaxf(cnt[i], 1.f);
}

// Pack fp32 W[K][N] -> bf16 Wt[N][KP] (transposed, K zero-padded to KP)
__global__ void pack_weight_kernel(const float* __restrict__ W,
                                   bf16_t* __restrict__ Wt, int K, int N,
                                   int KP) {
  int i = blockIdx.x * blockDim.x + threadIdx.x;
  if (i >= N * KP) return;
  int n = i / KP, k = i - n * KP;
  Wt[i] = (k < K) ? (bf16_t)W[(size_t)k * N + n] : (bf16_t)0.f;
}

// ---------------------------------------------------------------------------
// Input projection GVP: (s6, v3) -> (s128, v16), with activations. VALU-only
// (tiny: <1% of FLOPs). One block of 128 threads per node.
// ---------------------------------------------------------------------------
__global__ __launch_bounds__(128) void input_proj_kernel(
    const float* __restrict__ x_s, const float* __restrict__ x_v,
    const float* __restrict__ wh, const float* __restrict__ ws_b,
    const float* __restrict__ ws_w, const float* __restrict__ wv,
    float* __restrict__ sbuf, float* __restrict__ vbuf, int nNodes) {
  int n = blockIdx.x;
  if (n >= nNodes) return;
  int t = threadIdx.x;
  __shared__ float vh[16][3];
  __shared__ float vn[16];
  if (t < 48) {
    int h = t / 3, c = t - 3 * h;
    float a = 0.f;
    for (int k = 0; k < 3; ++k) a += x_v[n * 9 + k * 3 + c] * wh[k * 16 + h];
    vh[h][c] = a;
  }
  __syncthreads();
  if (t < 16) {
    float ss = vh[t][0] * vh[t][0] + vh[t][1] * vh[t][1] + vh[t][2] * vh[t][2];
    vn[t] = sqrtf(fmaxf(ss, EPSF));
  }
  __syncthreads();
  float acc = ws_b[t];
  for (int k = 0; k < 6; ++k) acc += x_s[n * 6 + k] * ws_w[k * 128 + t];
  for (int k = 0; k < 16; ++k) acc += vn[k] * ws_w[(6 + k) * 128 + t];
  sbuf[(size_t)n * 128 + t] = fmaxf(acc, 0.f);
  if (t < 16) {
    float a0 = 0.f, a1 = 0.f, a2 = 0.f;
    for (int h = 0; h < 16; ++h) {
      float w = wv[h * 16 + t];
      a0 += vh[h][0] * w;
      a1 += vh[h][1] * w;
      a2 += vh[h][2] * w;
    }
    float nn = sqrtf(fmaxf(a0 * a0 + a1 * a1 + a2 * a2, EPSF));
    float gt = 1.f / (1.f + expf(-nn));
    float* p = &vbuf[(size_t)n * 48 + 3 * t];
    p[0] = a0 * gt;
    p[1] = a1 * gt;
    p[2] = a2 * gt;
  }
}

// ---------------------------------------------------------------------------
// LayerNorm (+optional residual add with optional per-node scaling).
// s = LN(s + add_s*scale)*g + b ; v = (v + add_v*scale) / rms(||v_k||)
// One block of 128 threads per node.
// ---------------------------------------------------------------------------
__global__ __launch_bounds__(128) void ln_kernel(
    const float* __restrict__ s_in, const float* __restrict__ v_in,
    const float* __restrict__ add_s, const float* __restrict__ add_v,
    const float* __restrict__ inv_cnt, const float* __restrict__ g,
    const float* __restrict__ b, float* __restrict__ s_out,
    float* __restrict__ v_out, int nNodes) {
  int n = blockIdx.x;
  if (n >= nNodes) return;
  int t = threadIdx.x;
  __shared__ float red[128];
  __shared__ float vtmp[48];
  __shared__ float vs[17];
  float scale = inv_cnt ? inv_cnt[n] : 1.f;
  float x = s_in[(size_t)n * 128 + t];
  if (add_s) x += add_s[(size_t)n * 128 + t] * scale;
  red[t] = x;
  __syncthreads();
  for (int off = 64; off > 0; off >>= 1) {
    if (t < off) red[t] += red[t + off];
    __syncthreads();
  }
  float mu = red[0] * (1.f / 128.f);
  __syncthreads();
  float d = x - mu;
  red[t] = d * d;
  if (t < 48) {
    float vv = v_in[(size_t)n * 48 + t];
    if (add_v) vv += add_v[(size_t)n * 48 + t] * scale;
    vtmp[t] = vv;
  }
  __syncthreads();
  for (int off = 64; off > 0; off >>= 1) {
    if (t < off) red[t] += red[t + off];
    __syncthreads();
  }
  float var = red[0] * (1.f / 128.f);
  if (t < 16) {
    const float* p = &vtmp[3 * t];
    vs[t] = fmaxf(p[0] * p[0] + p[1] * p[1] + p[2] * p[2], EPSF);
  }
  __syncthreads();
  if (t == 0) {
    float m = 0.f;
    for (int i = 0; i < 16; ++i) m += vs[i];
    vs[16] = rsqrtf(m * (1.f / 16.f));
  }
  __syncthreads();
  s_out[(size_t)n * 128 + t] = d * rsqrtf(var + 1e-5f) * g[t] + b[t];
  if (t < 48) v_out[(size_t)n * 48 + t] = vtmp[t] * vs[16];
}

// ---------------------------------------------------------------------------
// Fused per-edge message pipeline: gather -> msg0 -> msg1 -> msg2 -> scatter.
// 16 edges per block, 256 threads (8 waves). Each wave owns a 16-channel
// N-tile of the 128 scalar outputs; ws_w GEMMs run on bf16 WMMA with f32 acc;
// the small vector-channel mixes (wh/wv, K<=33) run on fp32 VALU.
// ---------------------------------------------------------------------------
__global__ __launch_bounds__(256) void edge_msg_kernel(
    const float* __restrict__ sN, const float* __restrict__ vN,
    const int* __restrict__ srcI, const int* __restrict__ dstI,
    const float* __restrict__ eS, const float* __restrict__ eV,
    const bf16_t* __restrict__ Wt0, const float* __restrict__ b0,
    const float* __restrict__ wh0, const float* __restrict__ wv0,
    const bf16_t* __restrict__ Wt1, const float* __restrict__ b1,
    const float* __restrict__ wh1, const float* __restrict__ wv1,
    const bf16_t* __restrict__ Wt2, const float* __restrict__ b2,
    const float* __restrict__ wh2, const float* __restrict__ wv2,
    float* __restrict__ aggS, float* __restrict__ aggV, int nEdges) {
  __shared__ bf16_t A[16 * 352];   // activation tile (stage0: 321->352 cols)
  __shared__ float MV[16 * 99];    // vector features (<=33 vecs x 3)
  __shared__ float VH[16 * 99];    // vh (<=33 x 3)
  __shared__ int sIdx[16], dIdx[16];
  __shared__ unsigned char valid[16];

  const int t = threadIdx.x;
  const int lane = t & 31, wave = t >> 5, half = lane >> 4;
  const int e0 = blockIdx.x * 16;

  if (t < 16) {
    int eg = e0 + t;
    bool ok = eg < nEdges;
    valid[t] = ok ? 1 : 0;
    sIdx[t] = ok ? srcI[eg] : 0;
    dIdx[t] = ok ? dstI[eg] : 0;
  }
  __syncthreads();

  if (t == 0 && e0 + 16 < nEdges)
    __builtin_prefetch(eS + (size_t)(e0 + 16) * 32, 0, 1);

  // ---- gather scalar part: A[e][0..287] = [s_src(128) | edge_s(32) | s_dst]
  for (int idx = t; idx < 16 * 288; idx += 256) {
    int e = idx / 288, c = idx - e * 288;
    float val;
    if (c < 128)
      val = sN[(size_t)sIdx[e] * 128 + c];
    else if (c < 160)
      val = valid[e] ? eS[(size_t)(e0 + e) * 32 + (c - 128)] : 0.f;
    else
      val = sN[(size_t)dIdx[e] * 128 + (c - 160)];
    A[e * 352 + c] = (bf16_t)val;
  }
  for (int idx = t; idx < 16 * 31; idx += 256) {  // zero pad cols 321..351
    int e = idx / 31, c = 321 + (idx - e * 31);
    A[e * 352 + c] = (bf16_t)0.f;
  }
  // ---- gather vectors: MV[e][k][c]: v_src(16) | edge_v(1) | v_dst(16)
  for (int idx = t; idx < 16 * 99; idx += 256) {
    int e = idx / 99, r = idx - e * 99;
    int k = r / 3, c = r - 3 * k;
    float val;
    if (k < 16)
      val = vN[(size_t)sIdx[e] * 48 + k * 3 + c];
    else if (k == 16)
      val = valid[e] ? eV[(size_t)(e0 + e) * 3 + c] : 0.f;
    else
      val = vN[(size_t)dIdx[e] * 48 + (k - 17) * 3 + c];
    MV[idx] = val;
  }
  __syncthreads();

  // ================= stage 0 (msg0): si=288, vi=33, h=33 =================
  for (int idx = t; idx < 16 * 33; idx += 256) {
    int e = idx / 33, h = idx - e * 33;
    float a0 = 0.f, a1 = 0.f, a2 = 0.f;
    for (int k = 0; k < 33; ++k) {
      float w = wh0[k * 33 + h];
      const float* m = &MV[e * 99 + 3 * k];
      a0 += m[0] * w; a1 += m[1] * w; a2 += m[2] * w;
    }
    float* vh = &VH[e * 99 + 3 * h];
    vh[0] = a0; vh[1] = a1; vh[2] = a2;
    A[e * 352 + 288 + h] = (bf16_t)sqrtf(fmaxf(a0 * a0 + a1 * a1 + a2 * a2, EPSF));
  }
  __syncthreads();

  v8f acc = zero8();
  {
    const bf16_t* Bp = Wt0 + (size_t)(wave * 16) * 352;
#pragma unroll
    for (int kb = 0; kb < 11; ++kb) {
      v16bf af = load_frag16(A, 352, kb * 32, lane);
      v16bf bf = load_frag16(Bp, 352, kb * 32, lane);
      acc = wmma_bf16(af, bf, acc);
    }
  }
  float vox, voy, voz;
  {
    int e = t >> 4, o = t & 15;
    float a0 = 0.f, a1 = 0.f, a2 = 0.f;
    for (int h = 0; h < 33; ++h) {
      float w = wv0[h * 16 + o];
      const float* vh = &VH[e * 99 + 3 * h];
      a0 += vh[0] * w; a1 += vh[1] * w; a2 += vh[2] * w;
    }
    float nn = sqrtf(fmaxf(a0 * a0 + a1 * a1 + a2 * a2, EPSF));
    float gt = 1.f / (1.f + expf(-nn));
    vox = a0 * gt; voy = a1 * gt; voz = a2 * gt;
  }
  __syncthreads();
  {  // write stage-1 inputs: relu(s+bias) as bf16, gated vectors, zero pads
    int cb = wave * 16 + (lane & 15);
    float bias = b0[cb];
#pragma unroll
    for (int i = 0; i < 8; ++i) {
      int m = i + half * 8;
      A[m * 352 + cb] = (bf16_t)fmaxf(acc[i] + bias, 0.f);
    }
    int e = t >> 4, o = t & 15;
    float* mv = &MV[e * 99 + 3 * o];
    mv[0] = vox; mv[1] = voy; mv[2] = voz;
    A[e * 352 + 144 + o] = (bf16_t)0.f;  // pad cols 144..159
  }
  __syncthreads();

  // ================= stage 1 (msg1): si=128, vi=16, h=16 =================
  {
    int e = t >> 4, h = t & 15;
    float a0 = 0.f, a1 = 0.f, a2 = 0.f;
    for (int k = 0; k < 16; ++k) {
      float w = wh1[k * 16 + h];
      const float* m = &MV[e * 99 + 3 * k];
      a0 += m[0] * w; a1 += m[1] * w; a2 += m[2] * w;
    }
    float* vh = &VH[e * 99 + 3 * h];
    vh[0] = a0; vh[1] = a1; vh[2] = a2;
    A[e * 352 + 128 + h] = (bf16_t)sqrtf(fmaxf(a0 * a0 + a1 * a1 + a2 * a2, EPSF));
  }
  __syncthreads();
  acc = zero8();
  {
    const bf16_t* Bp = Wt1 + (size_t)(wave * 16) * 160;
#pragma unroll
    for (int kb = 0; kb < 5; ++kb) {
      v16bf af = load_frag16(A, 352, kb * 32, lane);
      v16bf bf = load_frag16(Bp, 160, kb * 32, lane);
      acc = wmma_bf16(af, bf, acc);
    }
  }
  {
    int e = t >> 4, o = t & 15;
    float a0 = 0.f, a1 = 0.f, a2 = 0.f;
    for (int h = 0; h < 16; ++h) {
      float w = wv1[h * 16 + o];
      const float* vh = &VH[e * 99 + 3 * h];
      a0 += vh[0] * w; a1 += vh[1] * w; a2 += vh[2] * w;
    }
    float nn = sqrtf(fmaxf(a0 * a0 + a1 * a1 + a2 * a2, EPSF));
    float gt = 1.f / (1.f + expf(-nn));
    vox = a0 * gt; voy = a1 * gt; voz = a2 * gt;
  }
  __syncthreads();
  {
    int cb = wave * 16 + (lane & 15);
    float bias = b1[cb];
#pragma unroll
    for (int i = 0; i < 8; ++i) {
      int m = i + half * 8;
      A[m * 352 + cb] = (bf16_t)fmaxf(acc[i] + bias, 0.f);
    }
    int e = t >> 4, o = t & 15;
    float* mv = &MV[e * 99 + 3 * o];
    mv[0] = vox; mv[1] = voy; mv[2] = voz;
  }
  __syncthreads();

  // ================= stage 2 (msg2): no activations =================
  {
    int e = t >> 4, h = t & 15;
    float a0 = 0.f, a1 = 0.f, a2 = 0.f;
    for (int k = 0; k < 16; ++k) {
      float w = wh2[k * 16 + h];
      const float* m = &MV[e * 99 + 3 * k];
      a0 += m[0] * w; a1 += m[1] * w; a2 += m[2] * w;
    }
    float* vh = &VH[e * 99 + 3 * h];
    vh[0] = a0; vh[1] = a1; vh[2] = a2;
    A[e * 352 + 128 + h] = (bf16_t)sqrtf(fmaxf(a0 * a0 + a1 * a1 + a2 * a2, EPSF));
  }
  __syncthreads();
  acc = zero8();
  {
    const bf16_t* Bp = Wt2 + (size_t)(wave * 16) * 160;
#pragma unroll
    for (int kb = 0; kb < 5; ++kb) {
      v16bf af = load_frag16(A, 352, kb * 32, lane);
      v16bf bf = load_frag16(Bp, 160, kb * 32, lane);
      acc = wmma_bf16(af, bf, acc);
    }
  }
  {
    int e = t >> 4, o = t & 15;
    float a0 = 0.f, a1 = 0.f, a2 = 0.f;
    for (int h = 0; h < 16; ++h) {
      float w = wv2[h * 16 + o];
      const float* vh = &VH[e * 99 + 3 * h];
      a0 += vh[0] * w; a1 += vh[1] * w; a2 += vh[2] * w;
    }
    vox = a0; voy = a1; voz = a2;  // no gate
  }
  // ---- scatter-add to destination nodes (mean applied later via inv_cnt)
  {
    int cb = wave * 16 + (lane & 15);
    float bias = b2[cb];
#pragma unroll
    for (int i = 0; i < 8; ++i) {
      int m = i + half * 8;
      if (valid[m]) atomicAdd(&aggS[(size_t)dIdx[m] * 128 + cb], acc[i] + bias);
    }
    int e = t >> 4, o = t & 15;
    if (valid[e]) {
      float* p = &aggV[(size_t)dIdx[e] * 48 + 3 * o];
      atomicAdd(&p[0], vox);
      atomicAdd(&p[1], voy);
      atomicAdd(&p[2], voz);
    }
  }
}

// ---------------------------------------------------------------------------
// Fused per-node feedforward: ff0 (128,16)->(512,32) act; ff1 (512,32)->(128,16).
// 16 nodes per block, 256 threads (8 waves). ff1 K = 544 = 17*32 exactly.
// ---------------------------------------------------------------------------
__global__ __launch_bounds__(256) void node_ff_kernel(
    const float* __restrict__ sN, const float* __restrict__ vN,
    const bf16_t* __restrict__ Wt0, const float* __restrict__ wh0,
    const float* __restrict__ wsb0, const float* __restrict__ wv0,
    const bf16_t* __restrict__ Wt1, const float* __restrict__ wh1,
    const float* __restrict__ wsb1, const float* __restrict__ wv1,
    float* __restrict__ ffS, float* __restrict__ ffV, int nNodes) {
  __shared__ bf16_t A[16 * 544];
  __shared__ float MV[16 * 96];
  __shared__ float VH[16 * 96];
  const int t = threadIdx.x;
  const int lane = t & 31, wave = t >> 5, half = lane >> 4;
  const int n0 = blockIdx.x * 16;

  for (int idx = t; idx < 16 * 128; idx += 256) {
    int n = idx >> 7, c = idx & 127, ng = n0 + n;
    A[n * 544 + c] = (bf16_t)((ng < nNodes) ? sN[(size_t)ng * 128 + c] : 0.f);
  }
  for (int idx = t; idx < 16 * 48; idx += 256) {
    int n = idx / 48, r = idx - 48 * n, ng = n0 + n;
    MV[n * 96 + r] = (ng < nNodes) ? vN[(size_t)ng * 48 + r] : 0.f;
  }
  __syncthreads();

  // ---- ff0: vh (h=32, K=16), vn -> A cols 128..159
  for (int idx = t; idx < 16 * 32; idx += 256) {
    int n = idx >> 5, h = idx & 31;
    float a0 = 0.f, a1 = 0.f, a2 = 0.f;
    for (int k = 0; k < 16; ++k) {
      float w = wh0[k * 32 + h];
      const float* m = &MV[n * 96 + 3 * k];
      a0 += m[0] * w; a1 += m[1] * w; a2 += m[2] * w;
    }
    float* vh = &VH[n * 96 + 3 * h];
    vh[0] = a0; vh[1] = a1; vh[2] = a2;
    A[n * 544 + 128 + h] = (bf16_t)sqrtf(fmaxf(a0 * a0 + a1 * a1 + a2 * a2, EPSF));
  }
  __syncthreads();

  v8f acc0[4];
#pragma unroll
  for (int tt = 0; tt < 4; ++tt) {
    acc0[tt] = zero8();
    const bf16_t* Bp = Wt0 + (size_t)(wave * 64 + tt * 16) * 160;
#pragma unroll
    for (int kb = 0; kb < 5; ++kb) {
      v16bf af = load_frag16(A, 544, kb * 32, lane);
      v16bf bf = load_frag16(Bp, 160, kb * 32, lane);
      acc0[tt] = wmma_bf16(af, bf, acc0[tt]);
    }
  }
  float vo[2][3];
#pragma unroll
  for (int it = 0; it < 2; ++it) {
    int idx = t + it * 256;
    int n = idx >> 5, o = idx & 31;
    float a0 = 0.f, a1 = 0.f, a2 = 0.f;
    for (int h = 0; h < 32; ++h) {
      float w = wv0[h * 32 + o];
      const float* vh = &VH[n * 96 + 3 * h];
      a0 += vh[0] * w; a1 += vh[1] * w; a2 += vh[2] * w;
    }
    float nn = sqrtf(fmaxf(a0 * a0 + a1 * a1 + a2 * a2, EPSF));
    float gt = 1.f / (1.f + expf(-nn));
    vo[it][0] = a0 * gt; vo[it][1] = a1 * gt; vo[it][2] = a2 * gt;
  }
  __syncthreads();
#pragma unroll
  for (int tt = 0; tt < 4; ++tt) {
    int cb = wave * 64 + tt * 16 + (lane & 15);
    float bias = wsb0[cb];
#pragma unroll
    for (int i = 0; i < 8; ++i) {
      int m = i + half * 8;
      A[m * 544 + cb] = (bf16_t)fmaxf(acc0[tt][i] + bias, 0.f);
    }
  }
#pragma unroll
  for (int it = 0; it < 2; ++it) {
    int idx = t + it * 256;
    int n = idx >> 5, o = idx & 31;
    float* mv = &MV[n * 96 + 3 * o];
    mv[0] = vo[it][0]; mv[1] = vo[it][1]; mv[2] = vo[it][2];
  }
  __syncthreads();

  // ---- ff1: vh (h=32, K=32), vn -> A cols 512..543
  for (int idx = t; idx < 16 * 32; idx += 256) {
    int n = idx >> 5, h = idx & 31;
    float a0 = 0.f, a1 = 0.f, a2 = 0.f;
    for (int k = 0; k < 32; ++k) {
      float w = wh1[k * 32 + h];
      const float* m = &MV[n * 96 + 3 * k];
      a0 += m[0] * w; a1 += m[1] * w; a2 += m[2] * w;
    }
    float* vh = &VH[n * 96 + 3 * h];
    vh[0] = a0; vh[1] = a1; vh[2] = a2;
    A[n * 544 + 512 + h] = (bf16_t)sqrtf(fmaxf(a0 * a0 + a1 * a1 + a2 * a2, EPSF));
  }
  __syncthreads();
  v8f acc = zero8();
  {
    const bf16_t* Bp = Wt1 + (size_t)(wave * 16) * 544;
#pragma unroll
    for (int kb = 0; kb < 17; ++kb) {
      v16bf af = load_frag16(A, 544, kb * 32, lane);
      v16bf bf = load_frag16(Bp, 544, kb * 32, lane);
      acc = wmma_bf16(af, bf, acc);
    }
  }
  float v1x, v1y, v1z;
  {
    int n = t >> 4, o = t & 15;
    float a0 = 0.f, a1 = 0.f, a2 = 0.f;
    for (int h = 0; h < 32; ++h) {
      float w = wv1[h * 16 + o];
      const float* vh = &VH[n * 96 + 3 * h];
      a0 += vh[0] * w; a1 += vh[1] * w; a2 += vh[2] * w;
    }
    v1x = a0; v1y = a1; v1z = a2;  // no activation on ff1
  }
  {
    int cb = wave * 16 + (lane & 15);
    float bias = wsb1[cb];
#pragma unroll
    for (int i = 0; i < 8; ++i) {
      int m = i + half * 8, ng = n0 + m;
      if (ng < nNodes) ffS[(size_t)ng * 128 + cb] = acc[i] + bias;
    }
    int n = t >> 4, o = t & 15, ng = n0 + n;
    if (ng < nNodes) {
      float* p = &ffV[(size_t)ng * 48 + 3 * o];
      p[0] = v1x; p[1] = v1y; p[2] = v1z;
    }
  }
}

// ---------------------------------------------------------------------------
// Host-side orchestration
// ---------------------------------------------------------------------------
extern "C" void kernel_launch(void* const* d_in, const int* in_sizes, int n_in,
                              void* d_out, int out_size, void* d_ws,
                              size_t ws_size, hipStream_t stream) {
  const int N = 20000, E = 320000;
  if (n_in < 83) return;  // unexpected param layout

  // d_in layout: setup_inputs() insertion order; 'params' leaves flattened in
  // jax pytree (alphabetical) order: final_ln{b,g}, input_proj{wh,ws_b,ws_w,wv},
  // layers[l]{ff0,ff1,ln0,ln1,msg0,msg1,msg2} each GVP = {wh,ws_b,ws_w,wv}.
  const float* x_s    = (const float*)d_in[0];
  const float* x_v    = (const float*)d_in[1];
  const int*   eidx   = (const int*)d_in[2];
  const float* edge_s = (const float*)d_in[3];
  const float* edge_v = (const float*)d_in[4];
  const float* fln_b  = (const float*)d_in[5];
  const float* fln_g  = (const float*)d_in[6];
  const float* ip_wh  = (const float*)d_in[7];
  const float* ip_wsb = (const float*)d_in[8];
  const float* ip_wsw = (const float*)d_in[9];
  const float* ip_wv  = (const float*)d_in[10];

  struct GVP { const float *wh, *wsb, *wsw, *wv; };
  struct LN  { const float *b, *g; };
  GVP ff0[3], ff1[3], msg0[3], msg1[3], msg2[3];
  LN  ln0[3], ln1[3];
  for (int l = 0; l < 3; ++l) {
    int bs = 11 + l * 24;
    ff0[l]  = {(const float*)d_in[bs + 0],  (const float*)d_in[bs + 1],
               (const float*)d_in[bs + 2],  (const float*)d_in[bs + 3]};
    ff1[l]  = {(const float*)d_in[bs + 4],  (const float*)d_in[bs + 5],
               (const float*)d_in[bs + 6],  (const float*)d_in[bs + 7]};
    ln0[l]  = {(const float*)d_in[bs + 8],  (const float*)d_in[bs + 9]};
    ln1[l]  = {(const float*)d_in[bs + 10], (const float*)d_in[bs + 11]};
    msg0[l] = {(const float*)d_in[bs + 12], (const float*)d_in[bs + 13],
               (const float*)d_in[bs + 14], (const float*)d_in[bs + 15]};
    msg1[l] = {(const float*)d_in[bs + 16], (const float*)d_in[bs + 17],
               (const float*)d_in[bs + 18], (const float*)d_in[bs + 19]};
    msg2[l] = {(const float*)d_in[bs + 20], (const float*)d_in[bs + 21],
               (const float*)d_in[bs + 22], (const float*)d_in[bs + 23]};
  }
  const int* srcP = eidx;
  const int* dstP = eidx + E;

  // Workspace carving
  char* wp = (char*)d_ws;
  auto carve = [&](size_t bytes) -> void* {
    void* p = (void*)wp;
    wp += (bytes + 255) & ~(size_t)255;
    return p;
  };
  float* sbuf = (float*)carve((size_t)N * 128 * 4);
  float* vbuf = (float*)carve((size_t)N * 48 * 4);
  float* aggS = (float*)carve((size_t)N * 128 * 4);
  float* aggV = (float*)carve((size_t)N * 48 * 4);
  float* icnt = (float*)carve((size_t)N * 4);
  bf16_t* WtM0[3];
  bf16_t* WtM1[3];
  bf16_t* WtM2[3];
  bf16_t* WtF0[3];
  bf16_t* WtF1[3];
  for (int l = 0; l < 3; ++l) {
    WtM0[l] = (bf16_t*)carve((size_t)128 * 352 * 2);
    WtM1[l] = (bf16_t*)carve((size_t)128 * 160 * 2);
    WtM2[l] = (bf16_t*)carve((size_t)128 * 160 * 2);
    WtF0[l] = (bf16_t*)carve((size_t)512 * 160 * 2);
    WtF1[l] = (bf16_t*)carve((size_t)128 * 544 * 2);
  }
  // agg buffers are free after ln0 consumes them -> reuse for ff outputs
  float* ffS = aggS;
  float* ffV = aggV;

  auto pack = [&](const float* W, bf16_t* Wt, int K, int Nn, int KP) {
    int tot = Nn * KP;
    pack_weight_kernel<<<(tot + 255) / 256, 256, 0, stream>>>(W, Wt, K, Nn, KP);
  };
  for (int l = 0; l < 3; ++l) {
    pack(msg0[l].wsw, WtM0[l], 321, 128, 352);
    pack(msg1[l].wsw, WtM1[l], 144, 128, 160);
    pack(msg2[l].wsw, WtM2[l], 144, 128, 160);
    pack(ff0[l].wsw,  WtF0[l], 160, 512, 160);
    pack(ff1[l].wsw,  WtF1[l], 544, 128, 544);
  }

  // counts -> inverse counts
  fill_zero_kernel<<<(N + 255) / 256, 256, 0, stream>>>(icnt, N);
  edge_count_kernel<<<(E + 255) / 256, 256, 0, stream>>>(dstP, icnt, E);
  inv_count_kernel<<<(N + 255) / 256, 256, 0, stream>>>(icnt, N);

  // input projection
  input_proj_kernel<<<N, 128, 0, stream>>>(x_s, x_v, ip_wh, ip_wsb, ip_wsw,
                                           ip_wv, sbuf, vbuf, N);

  for (int l = 0; l < 3; ++l) {
    fill_zero_kernel<<<((N * 128) + 255) / 256, 256, 0, stream>>>(aggS, N * 128);
    fill_zero_kernel<<<((N * 48) + 255) / 256, 256, 0, stream>>>(aggV, N * 48);

    edge_msg_kernel<<<(E + 15) / 16, 256, 0, stream>>>(
        sbuf, vbuf, srcP, dstP, edge_s, edge_v,
        WtM0[l], msg0[l].wsb, msg0[l].wh, msg0[l].wv,
        WtM1[l], msg1[l].wsb, msg1[l].wh, msg1[l].wv,
        WtM2[l], msg2[l].wsb, msg2[l].wh, msg2[l].wv,
        aggS, aggV, E);

    ln_kernel<<<N, 128, 0, stream>>>(sbuf, vbuf, aggS, aggV, icnt, ln0[l].g,
                                     ln0[l].b, sbuf, vbuf, N);

    node_ff_kernel<<<(N + 15) / 16, 256, 0, stream>>>(
        sbuf, vbuf,
        WtF0[l], ff0[l].wh, ff0[l].wsb, ff0[l].wv,
        WtF1[l], ff1[l].wh, ff1[l].wsb, ff1[l].wv,
        ffS, ffV, N);

    ln_kernel<<<N, 128, 0, stream>>>(sbuf, vbuf, ffS, ffV, nullptr, ln1[l].g,
                                     ln1[l].b, sbuf, vbuf, N);
  }

  float* out_s = (float*)d_out;
  float* out_v = out_s + (size_t)N * 128;
  ln_kernel<<<N, 128, 0, stream>>>(sbuf, vbuf, nullptr, nullptr, nullptr,
                                   fln_g, fln_b, out_s, out_v, N);
}
